// MultiHeadedSelfAttention_16484084482417
// MI455X (gfx1250) — compile-verified, matrix operand order checked by reading, not run on hardware
//
#include <hip/hip_runtime.h>
#include <hip/hip_bf16.h>

// ---------------------------------------------------------------------------
// MI455X (gfx1250) multi-head self-attention, bf16 WMMA path, fp32 accumulate.
// GEMM tiles staged by the Tensor Data Mover (tensor_load_to_lds, TENSORcnt);
// attention tiles staged by async global->LDS copies (ASYNCcnt).
// B=2, S=2048, D=1024, H=16, DK=64.
// ---------------------------------------------------------------------------

#define B_SZ 2
#define S_LEN 2048
#define DMODEL 1024
#define HEADS 16
#define DK 64
#define NEGV (-1000000000.0f)
#define SCALE 0.125f /* 1/sqrt(64) */

typedef __bf16 bf16_t;
typedef __attribute__((ext_vector_type(16))) __bf16 v16bf;
typedef __attribute__((ext_vector_type(8))) float v8f;
typedef unsigned __attribute__((ext_vector_type(4))) v4u;
typedef int __attribute__((ext_vector_type(4))) v4i_t;
typedef int __attribute__((ext_vector_type(8))) v8i_t;

#if defined(__AMDGCN__) && __has_builtin(__builtin_amdgcn_tensor_load_to_lds)
#define USE_TDM 1
#else
#define USE_TDM 0
#endif

__device__ __forceinline__ v8f wmma_bf16(v16bf a, v16bf b, v8f c) {
  return __builtin_amdgcn_wmma_f32_16x16x32_bf16(
      /*neg_a=*/false, a, /*neg_b=*/false, b,
      /*c_mod=*/(short)0, c, /*reuse_a=*/false, /*reuse_b=*/false);
}

// CDNA5 async global->LDS copy (16B per lane), tracked by ASYNCcnt.
__device__ __forceinline__ void async_copy_b128(unsigned lds_off, const void* gptr) {
  asm volatile("global_load_async_to_lds_b128 %0, %1, off"
               :: "v"(lds_off), "v"(gptr)
               : "memory");
}

__device__ __forceinline__ void wait_async0() {
#if __has_builtin(__builtin_amdgcn_s_wait_asynccnt)
  __builtin_amdgcn_s_wait_asynccnt(0);
#else
  asm volatile("s_wait_asynccnt 0" ::: "memory");
#endif
}
__device__ __forceinline__ void wait_async8() {  // allow 1 in-flight tile (8 copies/lane)
#if __has_builtin(__builtin_amdgcn_s_wait_asynccnt)
  __builtin_amdgcn_s_wait_asynccnt(8);
#else
  asm volatile("s_wait_asynccnt 8" ::: "memory");
#endif
}

#if USE_TDM
// Issue one 2D TDM tile load: tile_d1 rows x tile_d0 elements (2B each) from a
// row-major tensor with row stride stride0 (elements) into contiguous LDS.
// D# bit layout per CDNA5 ISA.
__device__ __forceinline__ void tdm_load_2d(unsigned lds_off, const void* gaddr,
                                            unsigned tensor_d0, unsigned tensor_d1,
                                            unsigned tile_d0, unsigned tile_d1,
                                            unsigned stride0) {
  unsigned long long ga = (unsigned long long)(uintptr_t)gaddr;
  v4u g0;
  g0.x = 0x1u;                                  // count=1 (valid), user mode
  g0.y = lds_off;                               // [63:32] lds_addr (bytes)
  g0.z = (unsigned)(ga & 0xFFFFFFFFu);          // [95:64] global_addr lo
  g0.w = (unsigned)((ga >> 32) & 0x01FFFFFFu)   // [120:96] global_addr hi
         | (2u << 30);                          // [127:126] type=2 ("image")
  v8i_t g1;
  g1[0] = (int)(1u << 16);                      // workgroup_mask=0, data_size=1 (2B)
  g1[1] = (int)((tensor_d0 & 0xFFFFu) << 16);   // [63:48] tensor_dim0 lo16
  g1[2] = (int)(((tensor_d1 & 0xFFFFu) << 16) | (tensor_d0 >> 16)); // [95:64]
  g1[3] = (int)(((tile_d0 & 0xFFFFu) << 16) | (tensor_d1 >> 16));   // [127:96]
  g1[4] = (int)(tile_d1 & 0xFFFFu);             // [143:128] tile_dim1, tile_dim2=0
  g1[5] = (int)stride0;                         // [191:160] dim0 stride lo32
  g1[6] = 0;                                    // stride0 hi16 | stride1 lo16
  g1[7] = 0;                                    // stride1 hi
  v4i_t z4 = {0, 0, 0, 0};
#if __clang_major__ >= 23
  v8i_t z8 = {0, 0, 0, 0, 0, 0, 0, 0};
  __builtin_amdgcn_tensor_load_to_lds(g0, g1, z4, z4, z8, 0);
#else
  __builtin_amdgcn_tensor_load_to_lds(g0, g1, z4, z4, 0);
#endif
}
#endif

// LDS byte offset of a shared-memory element.
#define LDS_OFF(p) ((unsigned)(uintptr_t)(p))

// Load a 16-bit A/B WMMA fragment from a K-contiguous row of 32 bf16.
// Lanes 0-15 hold K=0..7,16..23; lanes 16-31 hold K=8..15,24..31. kbase_=(lane>>4)*8.
#define LDFRAG(dst, rowptr, kbase_)                                          \
  do {                                                                       \
    union { v16bf v; int4 q[2]; } u_;                                        \
    u_.q[0] = *reinterpret_cast<const int4*>(&(rowptr)[(kbase_)]);           \
    u_.q[1] = *reinterpret_cast<const int4*>(&(rowptr)[(kbase_) + 16]);      \
    (dst) = u_.v;                                                            \
  } while (0)

// ---------------------------------------------------------------------------
__global__ __launch_bounds__(256) void cvt_f32_bf16(const float* __restrict__ in,
                                                    bf16_t* __restrict__ out, int n) {
  int i = blockIdx.x * 256 + threadIdx.x;
  if (i < n) out[i] = (bf16_t)in[i];
}

__global__ __launch_bounds__(256) void cvt_w_transpose(const float* __restrict__ W,
                                                       bf16_t* __restrict__ Wt) {
  int idx = blockIdx.x * 256 + threadIdx.x;  // over 1024*1024
  int k = idx >> 10, n = idx & 1023;
  Wt[((size_t)n << 10) | k] = (bf16_t)W[idx];
}

__global__ __launch_bounds__(256) void pack_mask(const int* __restrict__ mask,
                                                 unsigned* __restrict__ mp, int nwords) {
  int w = blockIdx.x * 8 + (threadIdx.x >> 5);
  int lane = threadIdx.x & 31;
  if (w < nwords) {
    int v = mask[(size_t)w * 32 + lane];
    unsigned long long bal = __ballot(v != 0);
    if (lane == 0) mp[w] = (unsigned)bal;
  }
}

// ---------------------------------------------------------------------------
// NT GEMM  C[4096][1024] = A[4096][1024] * Bt[1024][1024]^T + bias
// Block 256 threads = 8 waves (2 M x 4 N), block tile 128x128, K-step 64.
// Double-buffered LDS tiles filled by the TDM (2 descriptors per k-slice,
// issued by wave 0, TENSORcnt) or per-lane async copies as fallback.
// 16 WMMAs per barrier.
// ---------------------------------------------------------------------------
__global__ __launch_bounds__(256) void gemm_nt_bf16(const bf16_t* __restrict__ A,
                                                    const bf16_t* __restrict__ Bt,
                                                    const float* __restrict__ bias,
                                                    bf16_t* __restrict__ outBf,
                                                    float* __restrict__ outF32) {
  __shared__ __align__(16) bf16_t Asub[2][128][64];
  __shared__ __align__(16) bf16_t Bsub[2][128][64];

  const int tid = threadIdx.x;
  const int w = tid >> 5, lane = tid & 31;
  const int wm = w >> 2, wn = w & 3;  // 2 x 4 wave grid
  const int lm = lane & 15, hi = lane >> 4, kb = hi * 8;
  const int m0 = blockIdx.y * 128;
  const int n0 = blockIdx.x * 128;

  auto stage = [&](int k0, int buf) {
#if USE_TDM
    if (w == 0) {  // one TDM descriptor pair per k-slice; EXEC-independent op
      tdm_load_2d(LDS_OFF(&Asub[buf][0][0]), &A[(size_t)m0 * DMODEL + k0],
                  /*tensor_d0=*/DMODEL, /*tensor_d1=*/128,
                  /*tile_d0=*/64, /*tile_d1=*/128, /*stride0=*/DMODEL);
      tdm_load_2d(LDS_OFF(&Bsub[buf][0][0]), &Bt[(size_t)n0 * DMODEL + k0],
                  DMODEL, 128, 64, 128, DMODEL);
    }
#else
#pragma unroll
    for (int c = tid; c < 1024; c += 256) {  // 128 rows x 8 chunks, both tiles
      int r = c >> 3, o = c & 7;
      async_copy_b128(LDS_OFF(&Asub[buf][r][o * 8]),
                      &A[(size_t)(m0 + r) * DMODEL + k0 + o * 8]);
      async_copy_b128(LDS_OFF(&Bsub[buf][r][o * 8]),
                      &Bt[(size_t)(n0 + r) * DMODEL + k0 + o * 8]);
    }
#endif
  };
  auto stage_wait = [&](bool nextInFlight) {
#if USE_TDM
    if (nextInFlight) __builtin_amdgcn_s_wait_tensorcnt((short)2);
    else              __builtin_amdgcn_s_wait_tensorcnt((short)0);
#else
    if (nextInFlight) wait_async8();
    else              wait_async0();
#endif
  };

  v8f acc[4][2];
#pragma unroll
  for (int mt = 0; mt < 4; ++mt) { acc[mt][0] = {}; acc[mt][1] = {}; }

  stage(0, 0);
  stage_wait(false);
  __syncthreads();

  int buf = 0;
  for (int k0 = 0; k0 < DMODEL; k0 += 64) {
    const bool hasNext = (k0 + 64 < DMODEL);
    if (hasNext) stage(k0 + 64, buf ^ 1);  // DMA next slice during compute

#pragma unroll
    for (int kk = 0; kk < 2; ++kk) {
      v16bf bfr0, bfr1;
      LDFRAG(bfr0, (Bsub[buf][wn * 32 + lm] + kk * 32), kb);
      LDFRAG(bfr1, (Bsub[buf][wn * 32 + 16 + lm] + kk * 32), kb);
#pragma unroll
      for (int mt = 0; mt < 4; ++mt) {
        v16bf af;
        LDFRAG(af, (Asub[buf][wm * 64 + mt * 16 + lm] + kk * 32), kb);
        acc[mt][0] = wmma_bf16(af, bfr0, acc[mt][0]);
        acc[mt][1] = wmma_bf16(af, bfr1, acc[mt][1]);
      }
    }

    stage_wait(hasNext);  // current slice complete; next may stay in flight
    __syncthreads();
    buf ^= 1;
  }

  // Epilogue: C-frag = lane holds col n=lm, rows i + 8*hi.
#pragma unroll
  for (int mt = 0; mt < 4; ++mt)
#pragma unroll
    for (int nt = 0; nt < 2; ++nt) {
      int col = n0 + wn * 32 + nt * 16 + lm;
      float bb = bias[col];
#pragma unroll
      for (int i = 0; i < 8; ++i) {
        int row = m0 + wm * 64 + mt * 16 + i + 8 * hi;
        float v = acc[mt][nt][i] + bb;
        if (outF32) outF32[(size_t)row * DMODEL + col] = v;
        else        outBf[(size_t)row * DMODEL + col] = (bf16_t)v;
      }
    }
}

// ---------------------------------------------------------------------------
// Flash attention (online softmax), one block per (b, h, 128 q-rows).
// 8 waves; each wave owns 16 q-rows; Q frags resident in VGPRs.
// K chunk (32x64) async-staged to LDS once per block; V chunk staged
// transposed. Both double-buffered. 8 WMMAs per 32-key chunk per wave.
// ---------------------------------------------------------------------------
__global__ __launch_bounds__(256) void attn_kernel(const bf16_t* __restrict__ Q,
                                                   const bf16_t* __restrict__ Km,
                                                   const bf16_t* __restrict__ V,
                                                   const unsigned* __restrict__ maskP,
                                                   bf16_t* __restrict__ O) {
  __shared__ __align__(16) bf16_t Kc[2][32][64];     // K chunk [key][dk]
  __shared__ __align__(16) bf16_t Vt[2][64][32];     // V chunk transposed [dk][key]
  __shared__ __align__(16) bf16_t Pbuf[8][16][32];   // per-wave P re-layout scratch

  const int tid = threadIdx.x;
  const int w = tid >> 5, lane = tid & 31;
  const int lm = lane & 15, hi = lane >> 4, kb = hi * 8;
  const int b = blockIdx.z, h = blockIdx.y;
  const int qw = blockIdx.x * 128 + w * 16;

  auto stageK = [&](int kc, int buf) {
    int r = tid >> 3, o = tid & 7;
    async_copy_b128(LDS_OFF(&Kc[buf][r][o * 8]),
                    &Km[((size_t)(b * S_LEN + kc + r)) * DMODEL + h * DK + o * 8]);
  };
  auto loadV = [&](int kc) {
    int key = tid >> 3, d0 = (tid & 7) * 8;
    return *reinterpret_cast<const int4*>(
        &V[((size_t)(b * S_LEN + kc + key)) * DMODEL + h * DK + d0]);
  };
  auto storeVt = [&](int4 vq, int buf) {
    int key = tid >> 3, d0 = (tid & 7) * 8;
    union { int4 q; bf16_t e[8]; } u; u.q = vq;
#pragma unroll
    for (int j = 0; j < 8; ++j) Vt[buf][d0 + j][key] = u.e[j];
  };

  // Resident Q fragments (DK=64 -> two 32-wide k-steps).
  const bf16_t* qrow = Q + ((size_t)(b * S_LEN + qw + lm)) * DMODEL + h * DK;
  v16bf qa0, qa1;
  LDFRAG(qa0, qrow, kb);
  LDFRAG(qa1, qrow + 32, kb);

  v8f ofr[4];
  ofr[0] = {}; ofr[1] = {}; ofr[2] = {}; ofr[3] = {};
  float mrow[8], lrow[8];
#pragma unroll
  for (int i = 0; i < 8; ++i) { mrow[i] = -3.0e38f; lrow[i] = 0.0f; }

  stageK(0, 0);
  storeVt(loadV(0), 0);
  wait_async0();
  __syncthreads();

  int cur = 0;
  for (int kc = 0; kc < S_LEN; kc += 32) {
    const bool hasNext = (kc + 32 < S_LEN);
    int4 vnext = {};
    if (hasNext) {  // overlap next chunk's loads with this chunk's math
      stageK(kc + 32, cur ^ 1);
      vnext = loadV(kc + 32);
    }

    // Scores: two 16-key tiles, each = 2 chained WMMAs over DK=64.
    v8f s[2];
#pragma unroll
    for (int c = 0; c < 2; ++c) {
      v16bf kf0, kf1;
      LDFRAG(kf0, Kc[cur][c * 16 + lm], kb);
      LDFRAG(kf1, (Kc[cur][c * 16 + lm] + 32), kb);
      v8f z = {};
      z = wmma_bf16(qa0, kf0, z);
      z = wmma_bf16(qa1, kf1, z);
      s[c] = z;
    }

    unsigned mw[8];
#pragma unroll
    for (int i = 0; i < 8; ++i) {
      int row = qw + i + 8 * hi;
      mw[i] = maskP[(((size_t)(b * S_LEN + row)) << 6) | (unsigned)(kc >> 5)];
    }

    // Online softmax per row (row lives across a 16-lane half-group).
#pragma unroll
    for (int i = 0; i < 8; ++i) {
      float s0 = s[0][i] * SCALE, s1 = s[1][i] * SCALE;
      bool msk0 = ((mw[i] >> lm) & 1u) == 0u;
      bool msk1 = ((mw[i] >> (16 + lm)) & 1u) == 0u;
      if (msk0) s0 = NEGV;
      if (msk1) s1 = NEGV;
      float cm = fmaxf(s0, s1);
#pragma unroll
      for (int off = 1; off < 16; off <<= 1) cm = fmaxf(cm, __shfl_xor(cm, off));
      float mn = fmaxf(mrow[i], cm);
      float resc = __expf(mrow[i] - mn);
      float p0 = msk0 ? 0.0f : __expf(s0 - mn);
      float p1 = msk1 ? 0.0f : __expf(s1 - mn);
      float rs = p0 + p1;
#pragma unroll
      for (int off = 1; off < 16; off <<= 1) rs += __shfl_xor(rs, off);
      lrow[i] = lrow[i] * resc + rs;
      mrow[i] = mn;
#pragma unroll
      for (int t = 0; t < 4; ++t) ofr[t][i] = ofr[t][i] * resc;
      Pbuf[w][i + 8 * hi][lm]      = (bf16_t)p0;
      Pbuf[w][i + 8 * hi][16 + lm] = (bf16_t)p1;
    }

    // O += P(16x32) * V(32x64).
    v16bf pa;
    LDFRAG(pa, Pbuf[w][lm], kb);
#pragma unroll
    for (int t = 0; t < 4; ++t) {
      v16bf vb;
      LDFRAG(vb, Vt[cur][t * 16 + lm], kb);
      ofr[t] = wmma_bf16(pa, vb, ofr[t]);
    }

    if (hasNext) storeVt(vnext, cur ^ 1);
    wait_async0();
    __syncthreads();
    cur ^= 1;
  }

#pragma unroll
  for (int i = 0; i < 8; ++i) {
    float l = lrow[i];
    float inv = (l > 0.0f) ? 1.0f / l : 0.0f;  // fully-masked row -> zeros
    int row = qw + i + 8 * hi;
#pragma unroll
    for (int t = 0; t < 4; ++t) {
      O[((size_t)(b * S_LEN + row)) * DMODEL + h * DK + t * 16 + lm] =
          (bf16_t)(ofr[t][i] * inv);
    }
  }
}

// ---------------------------------------------------------------------------
extern "C" void kernel_launch(void* const* d_in, const int* in_sizes, int n_in,
                              void* d_out, int out_size, void* d_ws, size_t ws_size,
                              hipStream_t stream) {
  const float* x    = (const float*)d_in[0];
  const int*   mask = (const int*)d_in[1];
  const float* Wq   = (const float*)d_in[2];
  const float* bq   = (const float*)d_in[3];
  const float* Wk   = (const float*)d_in[4];
  const float* bk   = (const float*)d_in[5];
  const float* Wv   = (const float*)d_in[6];
  const float* bv   = (const float*)d_in[7];
  const float* Wo   = (const float*)d_in[8];
  const float* bo   = (const float*)d_in[9];
  float* out = (float*)d_out;

  char* ws = (char*)d_ws;
  const size_t MN = (size_t)B_SZ * S_LEN * DMODEL;  // 4096*1024
  bf16_t* xbf  = (bf16_t*)(ws + 0x0000000);         // 8 MB
  bf16_t* Wqt  = (bf16_t*)(ws + 0x0800000);         // 2 MB each
  bf16_t* Wkt  = (bf16_t*)(ws + 0x0A00000);
  bf16_t* Wvt  = (bf16_t*)(ws + 0x0C00000);
  bf16_t* Wot  = (bf16_t*)(ws + 0x0E00000);
  bf16_t* Qb   = (bf16_t*)(ws + 0x1000000);         // 8 MB each
  bf16_t* Kb   = (bf16_t*)(ws + 0x1800000);
  bf16_t* Vb   = (bf16_t*)(ws + 0x2000000);
  bf16_t* Ab   = (bf16_t*)(ws + 0x2800000);
  unsigned* mP = (unsigned*)(ws + 0x3000000);       // 1 MB

  cvt_f32_bf16<<<(int)(MN / 256), 256, 0, stream>>>(x, xbf, (int)MN);
  cvt_w_transpose<<<4096, 256, 0, stream>>>(Wq, Wqt);
  cvt_w_transpose<<<4096, 256, 0, stream>>>(Wk, Wkt);
  cvt_w_transpose<<<4096, 256, 0, stream>>>(Wv, Wvt);
  cvt_w_transpose<<<4096, 256, 0, stream>>>(Wo, Wot);

  const int nwords = B_SZ * S_LEN * (S_LEN / 32);
  pack_mask<<<nwords / 8, 256, 0, stream>>>(mask, mP, nwords);

  dim3 ggrid(DMODEL / 128, (B_SZ * S_LEN) / 128);
  gemm_nt_bf16<<<ggrid, 256, 0, stream>>>(xbf, Wqt, bq, Qb, nullptr);
  gemm_nt_bf16<<<ggrid, 256, 0, stream>>>(xbf, Wkt, bk, Kb, nullptr);
  gemm_nt_bf16<<<ggrid, 256, 0, stream>>>(xbf, Wvt, bv, Vb, nullptr);

  dim3 agrid(S_LEN / 128, HEADS, B_SZ);
  attn_kernel<<<agrid, 256, 0, stream>>>(Qb, Kb, Vb, mP, Ab);

  gemm_nt_bf16<<<ggrid, 256, 0, stream>>>(Ab, Wot, bo, nullptr, out);
}
